// NeuralNetwork_82540681494872
// MI455X (gfx1250) — matrix operand order repaired and checked by reading, not compiled, not run on hardware
//
#include <hip/hip_runtime.h>

typedef __attribute__((ext_vector_type(16))) _Float16 v16h;
typedef __attribute__((ext_vector_type(8)))  _Float16 half8;
typedef __attribute__((ext_vector_type(8)))  float    v8f;
typedef __attribute__((ext_vector_type(4)))  unsigned int u32x4;
typedef __attribute__((ext_vector_type(8)))  int i32x8;
typedef __attribute__((ext_vector_type(4)))  int i32x4;

union V16U { v16h v; half8 h[2]; };

#define THREADS 256
#define SLABS 4

// Node v in {0..5}: triangle mates m1,m2 (same triple), partner p=(v+3)%6.
__device__ __forceinline__ void trio(int v, int& m1, int& m2, int& p) {
  int tb = (v < 3) ? 0 : 3;
  int lv = v - tb;
  m1 = tb + ((lv + 1) % 3);
  m2 = tb + ((lv + 2) % 3);
  p  = (v + 3) % 6;
}

// Tensor Data Mover: 1-D DMA of n8*8 bytes global -> LDS (ISA 8.3-8.6 D# layout).
// Issue from one wave only; pair with s_wait_tensorcnt + workgroup barrier.
__device__ __forceinline__ void tdm_load_1d(const void* gsrc, const void* lds_dst,
                                            unsigned n8 /* 8-byte units, <= 65535 */) {
  unsigned long long ga = (unsigned long long)gsrc;
  unsigned lds = (unsigned)(unsigned long long)lds_dst;  // flat LDS addr: low 32 bits
  u32x4 g0;
  g0[0] = 1u;                                            // count=1 (valid), user mode
  g0[1] = lds;                                           // lds_addr (bytes)
  g0[2] = (unsigned)ga;                                  // global_addr[31:0]
  g0[3] = (unsigned)((ga >> 32) & 0x1FFFFFFu) | (2u << 30);  // addr[56:32] | type=2
  i32x8 g1;
  g1[0] = (int)(3u << 16);                               // wg_mask=0, data_size=8B
  g1[1] = (int)((n8 & 0xFFFFu) << 16);                   // tensor_dim0[15:0]
  g1[2] = (int)(((n8 >> 16) & 0xFFFFu) | (1u << 16));    // tensor_dim0[31:16] | tensor_dim1=1
  g1[3] = (int)((n8 & 0xFFFFu) << 16);                   // tile_dim0 = n8
  g1[4] = 1;                                             // tile_dim1 = 1, tile_dim2 = 0
  g1[5] = (int)n8;                                       // tensor_dim0_stride[31:0]
  g1[6] = 0;                                             // stride hi / dim1_stride lo
  g1[7] = 0;
  i32x4 z4 = {0, 0, 0, 0};
  i32x8 z8 = {0, 0, 0, 0, 0, 0, 0, 0};
  __builtin_amdgcn_tensor_load_to_lds(g0, g1, z4, z4, z8, 0);
}

// One wave computes 3 Mtiles x 1 Ntile of C += A * Wt^T using v_wmma_f32_16x16x32_f16.
// A: [rows][KD] f16 (row-major, LDS), Wt: [cols][KD] f16 (col-of-W major, LDS).
template<int KD>
__device__ __forceinline__ void gemm_tiles(const _Float16* A, const _Float16* Wt,
                                           int mbase, int ntile, int lane, v8f acc[3]) {
  const int r = lane & 15, hi = lane >> 4;
  #pragma unroll
  for (int k0 = 0; k0 < KD; k0 += 32) {
    V16U b;
    const _Float16* bp = Wt + (ntile * 16 + r) * KD + k0 + hi * 16;
    b.h[0] = *(const half8*)(bp);
    b.h[1] = *(const half8*)(bp + 8);
    #pragma unroll
    for (int mt = 0; mt < 3; ++mt) {
      V16U a;
      const _Float16* ap = A + ((mbase + mt) * 16 + r) * KD + k0 + hi * 16;
      a.h[0] = *(const half8*)(ap);
      a.h[1] = *(const half8*)(ap + 8);
      acc[mt] = __builtin_amdgcn_wmma_f32_16x16x32_f16(false, a.v, false, b.v,
                                                       (short)0, acc[mt], false, false);
    }
  }
}

// C/D layout: VGPR q holds row M = q + 8*(lane>=16), col N = lane&15.
template<int DOUT>
__device__ __forceinline__ void store_tiles(float* C, int mbase, int ntile, int lane,
                                            const v8f acc[3]) {
  const int r = lane & 15, hi = lane >> 4;
  #pragma unroll
  for (int mt = 0; mt < 3; ++mt)
    #pragma unroll
    for (int q = 0; q < 8; ++q)
      C[((mbase + mt) * 16 + q + 8 * hi) * DOUT + ntile * 16 + r] = acc[mt][q];
}

// Build mixed activations A[lr][0:128]=self, [128:256]=mate1+mate2, [256:384]=partner.
template<int ROWS>
__device__ __forceinline__ void build_mixed128(_Float16* A, const _Float16* src, int tid) {
  constexpr int CH = 48;  // 384/8 half8-chunks per row
  for (int idx = tid; idx < ROWS * CH; idx += THREADS) {
    int lr = idx / CH, ck = idx % CH;
    int sect = ck / 16;
    int kk = (ck % 16) * 8;
    int s6 = (lr / 6) * 6, v = lr % 6;
    int m1, m2, p; trio(v, m1, m2, p);
    half8 val;
    if (sect == 0)      val = *(const half8*)(src + (s6 + v)  * 128 + kk);
    else if (sect == 1) val = *(const half8*)(src + (s6 + m1) * 128 + kk)
                            + *(const half8*)(src + (s6 + m2) * 128 + kk);
    else                val = *(const half8*)(src + (s6 + p)  * 128 + kk);
    *(half8*)(A + lr * 384 + ck * 8) = val;
  }
}

// bias-add + LayerNorm + gamma/beta + ReLU, f32 in, f16 out.
// sB: [0:D)=bs+2b0+b1, [D:2D)=g, [2D:3D)=b.
template<int DOUT>
__device__ __forceinline__ void ln_relu(const float* C, const float* sB,
                                        _Float16* dst, int rows, int dstride, int tid) {
  for (int row = tid; row < rows; row += THREADS) {
    const float* cr = C + row * DOUT;
    float s0 = 0.f, s1 = 0.f;
    for (int c = 0; c < DOUT; ++c) { float t = cr[c] + sB[c]; s0 += t; s1 += t * t; }
    const float mu = s0 * (1.f / DOUT);
    const float var = s1 * (1.f / DOUT) - mu * mu;
    const float inv = rsqrtf(var + 1e-5f);
    _Float16* dr = dst + row * dstride;
    for (int c = 0; c < DOUT; ++c) {
      float y = (cr[c] + sB[c] - mu) * inv * sB[DOUT + c] + sB[2 * DOUT + c];
      dr[c] = (_Float16)(y > 0.f ? y : 0.f);
    }
  }
}

// Transpose-concat pack: dst[n*kpad + k] = {Ws,W0,W1}[k][n] as f16, zero-padded to kpad.
__global__ void pack_kernel(const float* __restrict__ Ws, const float* __restrict__ W0,
                            const float* __restrict__ W1, _Float16* __restrict__ dst,
                            int din, int dout, int kpad) {
  int total = dout * kpad;
  for (int idx = blockIdx.x * blockDim.x + threadIdx.x; idx < total;
       idx += gridDim.x * blockDim.x) {
    int n = idx / kpad, k = idx % kpad;
    float v = 0.f;
    if (k < din)          v = Ws[k * dout + n];
    else if (k < 2 * din) v = W0[(k - din) * dout + n];
    else if (k < 3 * din) v = W1[(k - 2 * din) * dout + n];
    dst[idx] = (_Float16)v;
  }
}

// x -> shared layer 1 -> shared layer 2 -> h2 (f16).
// Block: weights resident in LDS (via TDM), SLABS slabs x 8 samples (48 nodes).
__global__ __launch_bounds__(THREADS) void shared_kernel(
    const float* __restrict__ x,
    const _Float16* __restrict__ gW,     // Wt1 (128*32) immediately followed by Wt2 (128*384)
    const float* __restrict__ bs1, const float* __restrict__ b01, const float* __restrict__ b11,
    const float* __restrict__ g1,  const float* __restrict__ be1,
    const float* __restrict__ bs2, const float* __restrict__ b02, const float* __restrict__ b12,
    const float* __restrict__ g2,  const float* __restrict__ be2,
    _Float16* __restrict__ h2out) {
  constexpr int ROWS = 48;
  constexpr int NW = 128 * 32 + 128 * 384;
  __shared__ _Float16 sW[NW];                                // Wt1 | Wt2
  __shared__ __align__(16) unsigned char sU[ROWS * 384 * 2]; // A1 / A2 / C union
  __shared__ _Float16 sH1[ROWS * 128];
  __shared__ float sB1[3 * 128];
  __shared__ float sB2[3 * 128];

  const int tid = threadIdx.x, lane = tid & 31, w = tid >> 5;
  const _Float16* sWt1 = sW;
  const _Float16* sWt2 = sW + 128 * 32;

  if (w == 0) tdm_load_1d(gW, sW, NW * 2 / 8);   // async DMA weights -> LDS
  for (int i = tid; i < 128; i += THREADS) {
    sB1[i] = bs1[i] + 2.f * b01[i] + b11[i]; sB1[128 + i] = g1[i]; sB1[256 + i] = be1[i];
    sB2[i] = bs2[i] + 2.f * b02[i] + b12[i]; sB2[128 + i] = g2[i]; sB2[256 + i] = be2[i];
  }

  for (int slab = 0; slab < SLABS; ++slab) {
    const int sbase = (blockIdx.x * SLABS + slab) * 8;  // first sample of slab

    // Mixed layer-1 input A1 [48][32] (true K=12, padded with zeros).
    // h0[n*6 + a*3 + c][f] = x[n][f][a][c]  =>  per-sample floats xs[f*6 + v].
    _Float16* A1 = (_Float16*)sU;
    for (int idx = tid; idx < ROWS * 32; idx += THREADS) {
      int lr = idx >> 5, k = idx & 31;
      int s = lr / 6, v = lr % 6;
      const float* xs = x + (size_t)(sbase + s) * 24;
      float fv = 0.f;
      if (k < 12) {
        int sect = k >> 2, f = k & 3;
        int m1, m2, p; trio(v, m1, m2, p);
        if (sect == 0)      fv = xs[f * 6 + v];
        else if (sect == 1) fv = xs[f * 6 + m1] + xs[f * 6 + m2];
        else                fv = xs[f * 6 + p];
      }
      A1[lr * 32 + k] = (_Float16)fv;
    }
    if (slab == 0 && w == 0) __builtin_amdgcn_s_wait_tensorcnt(0);  // weights arrived
    __syncthreads();

    // Layer 1: [48 x 32] x [32 x 128]
    v8f zf = {};
    v8f acc[3]; acc[0] = zf; acc[1] = zf; acc[2] = zf;
    gemm_tiles<32>(A1, sWt1, 0, w, lane, acc);
    __syncthreads();  // all waves done reading A1 before C overwrites sU
    float* C = (float*)sU;
    store_tiles<128>(C, 0, w, lane, acc);
    __syncthreads();
    ln_relu<128>(C, sB1, sH1, ROWS, 128, tid);
    __syncthreads();

    // Layer 2: mix h1 -> [48 x 384] x [384 x 128]
    _Float16* A2 = (_Float16*)sU;
    build_mixed128<ROWS>(A2, sH1, tid);
    __syncthreads();
    acc[0] = zf; acc[1] = zf; acc[2] = zf;
    gemm_tiles<384>(A2, sWt2, 0, w, lane, acc);
    __syncthreads();
    store_tiles<128>(C, 0, w, lane, acc);
    __syncthreads();
    ln_relu<128>(C, sB2, h2out + (size_t)(blockIdx.x * SLABS + slab) * ROWS * 128,
                 ROWS, 128, tid);
    __syncthreads();  // protect sU before next slab's A1 build
  }
}

// h2 -> full(hid) (WMMA) -> output pel (VALU GEMV) -> d_out slice.
// Requires (DOUT/16) * (ROWS/16) == 24 (3 tiles per wave).
template<int DOUT, int NY, int ROWS>
__global__ __launch_bounds__(THREADS) void head_kernel(
    const _Float16* __restrict__ h2, const _Float16* __restrict__ gWt,
    const float* __restrict__ bs, const float* __restrict__ b0, const float* __restrict__ b1,
    const float* __restrict__ g,  const float* __restrict__ be,
    const float* __restrict__ oWs, const float* __restrict__ oW0, const float* __restrict__ oW1,
    const float* __restrict__ obs, const float* __restrict__ ob0, const float* __restrict__ ob1,
    float* __restrict__ outp) {
  constexpr int K3 = 384;
  constexpr int NT = DOUT / 16;
  __shared__ _Float16 sWt[DOUT * K3];
  __shared__ __align__(16) unsigned char sU[ROWS * K3 * 2];  // A / C union
  __shared__ _Float16 sT[ROWS * DOUT];
  __shared__ float sB[3 * DOUT];
  __shared__ float sOW[3 * DOUT * NY + NY];

  const int tid = threadIdx.x, lane = tid & 31, w = tid >> 5;
  const _Float16* srcbase = h2 + (size_t)blockIdx.x * SLABS * ROWS * 128;

  if (w == 0) tdm_load_1d(gWt, sWt, DOUT * K3 * 2 / 8);  // async DMA weights -> LDS
  for (int i = tid; i < DOUT; i += THREADS) {
    sB[i] = bs[i] + 2.f * b0[i] + b1[i]; sB[DOUT + i] = g[i]; sB[2 * DOUT + i] = be[i];
  }
  for (int i = tid; i < DOUT * NY; i += THREADS) {
    sOW[i] = oWs[i]; sOW[DOUT * NY + i] = oW0[i]; sOW[2 * DOUT * NY + i] = oW1[i];
  }
  if (tid < NY) sOW[3 * DOUT * NY + tid] = obs[tid] + 2.f * ob0[tid] + ob1[tid];

  for (int slab = 0; slab < SLABS; ++slab) {
    // Prefetch next slab's activations into cache while this slab computes.
    if (slab + 1 < SLABS && tid < ROWS * 2)
      __builtin_prefetch(srcbase + (size_t)(slab + 1) * ROWS * 128 + tid * 64, 0, 1);

    _Float16* A = (_Float16*)sU;
    build_mixed128<ROWS>(A, srcbase + (size_t)slab * ROWS * 128, tid);
    if (slab == 0 && w == 0) __builtin_amdgcn_s_wait_tensorcnt(0);
    __syncthreads();

    v8f zf = {};
    v8f acc[3]; acc[0] = zf; acc[1] = zf; acc[2] = zf;
    const int nt = w % NT, mbase = (w / NT) * 3;
    gemm_tiles<K3>(A, sWt, mbase, nt, lane, acc);
    __syncthreads();
    float* C = (float*)sU;
    store_tiles<DOUT>(C, mbase, nt, lane, acc);
    __syncthreads();
    ln_relu<DOUT>(C, sB, sT, ROWS, DOUT, tid);
    __syncthreads();

    // Output pel: out[lr][y] = t_self.Ws + (t_m1+t_m2).W0 + t_p.W1 + bias
    for (int idx = tid; idx < ROWS * NY; idx += THREADS) {
      int lr = idx / NY, y = idx % NY;
      int s6 = (lr / 6) * 6, v = lr % 6;
      int m1, m2, p; trio(v, m1, m2, p);
      const _Float16* t0 = sT + (s6 + v)  * DOUT;
      const _Float16* ta = sT + (s6 + m1) * DOUT;
      const _Float16* tb = sT + (s6 + m2) * DOUT;
      const _Float16* tp = sT + (s6 + p)  * DOUT;
      float a = sOW[3 * DOUT * NY + y];
      for (int k = 0; k < DOUT; k += 8) {
        half8 h0 = *(const half8*)(t0 + k);
        half8 ha = *(const half8*)(ta + k);
        half8 hb = *(const half8*)(tb + k);
        half8 hp = *(const half8*)(tp + k);
        #pragma unroll
        for (int j = 0; j < 8; ++j) {
          a += (float)h0[j] * sOW[(k + j) * NY + y]
             + ((float)ha[j] + (float)hb[j]) * sOW[DOUT * NY + (k + j) * NY + y]
             + (float)hp[j] * sOW[2 * DOUT * NY + (k + j) * NY + y];
        }
      }
      outp[(size_t)((blockIdx.x * SLABS + slab) * ROWS + lr) * NY + y] = a;
    }
    __syncthreads();  // sT/sU reuse next slab
  }
}

extern "C" void kernel_launch(void* const* d_in, const int* in_sizes, int n_in,
                              void* d_out, int out_size, void* d_ws, size_t ws_size,
                              hipStream_t stream) {
  (void)n_in; (void)out_size; (void)ws_size;
  const int NSmp = 32768, N = NSmp * 6;

  enum {
    X,
    S0_WS, S0_BS, S0_W0, S0_B0, S0_W1, S0_B1, S0_G, S0_B,
    S1_WS, S1_BS, S1_W0, S1_B0, S1_W1, S1_B1, S1_G, S1_B,
    STH_WS, STH_BS, STH_W0, STH_B0, STH_W1, STH_B1, STH_G, STH_B,
    STO_WS, STO_BS, STO_W0, STO_B0, STO_W1, STO_B1,
    GRH_WS, GRH_BS, GRH_W0, GRH_B0, GRH_W1, GRH_B1, GRH_G, GRH_B,
    GRO_WS, GRO_BS, GRO_W0, GRO_B0, GRO_W1, GRO_B1,
    F4H_WS, F4H_BS, F4H_W0, F4H_B0, F4H_W1, F4H_B1, F4H_G, F4H_B,
    F4O_WS, F4O_BS, F4O_W0, F4O_B0, F4O_W1, F4O_B1,
    NSLOT
  };
  int map[NSLOT];
  if (in_sizes[0] == NSmp * 24) {
    // Insertion-order flatten: enum above matches exactly.
    for (int i = 0; i < NSLOT; ++i) map[i] = i;
  } else {
    // jax pytree flatten: dict keys sorted; pel keys W0<W1<Ws<b0<b1<bs; layer keys b<g<pel.
    int j = 0;
    map[F4H_B]=j++; map[F4H_G]=j++; map[F4H_W0]=j++; map[F4H_W1]=j++; map[F4H_WS]=j++;
    map[F4H_B0]=j++; map[F4H_B1]=j++; map[F4H_BS]=j++;
    map[F4O_W0]=j++; map[F4O_W1]=j++; map[F4O_WS]=j++; map[F4O_B0]=j++; map[F4O_B1]=j++; map[F4O_BS]=j++;
    map[GRH_B]=j++; map[GRH_G]=j++; map[GRH_W0]=j++; map[GRH_W1]=j++; map[GRH_WS]=j++;
    map[GRH_B0]=j++; map[GRH_B1]=j++; map[GRH_BS]=j++;
    map[GRO_W0]=j++; map[GRO_W1]=j++; map[GRO_WS]=j++; map[GRO_B0]=j++; map[GRO_B1]=j++; map[GRO_BS]=j++;
    map[S0_B]=j++; map[S0_G]=j++; map[S0_W0]=j++; map[S0_W1]=j++; map[S0_WS]=j++;
    map[S0_B0]=j++; map[S0_B1]=j++; map[S0_BS]=j++;
    map[S1_B]=j++; map[S1_G]=j++; map[S1_W0]=j++; map[S1_W1]=j++; map[S1_WS]=j++;
    map[S1_B0]=j++; map[S1_B1]=j++; map[S1_BS]=j++;
    map[STH_B]=j++; map[STH_G]=j++; map[STH_W0]=j++; map[STH_W1]=j++; map[STH_WS]=j++;
    map[STH_B0]=j++; map[STH_B1]=j++; map[STH_BS]=j++;
    map[STO_W0]=j++; map[STO_W1]=j++; map[STO_WS]=j++; map[STO_B0]=j++; map[STO_B1]=j++; map[STO_BS]=j++;
    map[X]=j++;
  }
  auto F = [&](int slot) { return (const float*)d_in[map[slot]]; };

  // Workspace layout (f16 elements): packed transposed weights, then h2 activations.
  _Float16* wsh = (_Float16*)d_ws;
  _Float16* Wt1 = wsh;                    //  128*32
  _Float16* Wt2 = wsh + 4096;             //  128*384 (contiguous with Wt1)
  _Float16* Wst = wsh + 53248;            //   64*384
  _Float16* Wgr = wsh + 77824;            //   64*384
  _Float16* Wf4 = wsh + 102400;           //  128*384
  _Float16* H2  = wsh + 151552;           //  N*128

  pack_kernel<<<16,  THREADS, 0, stream>>>(F(S0_WS),  F(S0_W0),  F(S0_W1),  Wt1, 4,   128, 32);
  pack_kernel<<<192, THREADS, 0, stream>>>(F(S1_WS),  F(S1_W0),  F(S1_W1),  Wt2, 128, 128, 384);
  pack_kernel<<<96,  THREADS, 0, stream>>>(F(STH_WS), F(STH_W0), F(STH_W1), Wst, 128, 64,  384);
  pack_kernel<<<96,  THREADS, 0, stream>>>(F(GRH_WS), F(GRH_W0), F(GRH_W1), Wgr, 128, 64,  384);
  pack_kernel<<<192, THREADS, 0, stream>>>(F(F4H_WS), F(F4H_W0), F(F4H_W1), Wf4, 128, 128, 384);

  shared_kernel<<<1024, THREADS, 0, stream>>>(
      F(X), Wt1,
      F(S0_BS), F(S0_B0), F(S0_B1), F(S0_G), F(S0_B),
      F(S1_BS), F(S1_B0), F(S1_B1), F(S1_G), F(S1_B), H2);

  float* out = (float*)d_out;
  head_kernel<128, 4, 48><<<1024, THREADS, 0, stream>>>(
      H2, Wf4,
      F(F4H_BS), F(F4H_B0), F(F4H_B1), F(F4H_G), F(F4H_B),
      F(F4O_WS), F(F4O_W0), F(F4O_W1), F(F4O_BS), F(F4O_B0), F(F4O_B1),
      out);                               // x_f4: [N,4]
  head_kernel<64, 1, 96><<<512, THREADS, 0, stream>>>(
      H2, Wgr,
      F(GRH_BS), F(GRH_B0), F(GRH_B1), F(GRH_G), F(GRH_B),
      F(GRO_WS), F(GRO_W0), F(GRO_W1), F(GRO_BS), F(GRO_B0), F(GRO_B1),
      out + (size_t)N * 4);               // x_gr: [N,1]
  head_kernel<64, 1, 96><<<512, THREADS, 0, stream>>>(
      H2, Wst,
      F(STH_BS), F(STH_B0), F(STH_B1), F(STH_G), F(STH_B),
      F(STO_WS), F(STO_W0), F(STO_W1), F(STO_BS), F(STO_B0), F(STO_B1),
      out + (size_t)N * 5);               // x_st: [N,1]
}
